// RSNN_38878043963928
// MI455X (gfx1250) — matrix operand order repaired and checked by reading, not compiled, not run on hardware
//
#include <hip/hip_runtime.h>
#include <hip/hip_bf16.h>

typedef __attribute__((ext_vector_type(16))) _Float16 v16h;
typedef __attribute__((ext_vector_type(8)))  _Float16 v8h;
typedef __attribute__((ext_vector_type(8)))  float    v8f;

// ---- model constants ----
#define NN    2000
#define II    200
#define BB    32
#define TT    500
#define NEXC  1700
#define NEXCI 170
#define KP    2048   // N padded to power of two: 64 k-chunks of 32
#define IP    224    // I padded to multiple of 32 (7 k-chunks)
#define RING  32     // ring-buffer slots (>= 20 delay), power of 2
#define NTILE 125    // 2000/16 output-column tiles
#define KSEG  4      // K-split per output tile
#define CSEG  16     // 64 chunks / 4 segments: compile-time trip count
#define GRIDB 125    // blocks: 2 tiles * 4 segs = 8 waves per block

#define DECAY_E 0.951229424500714f   // exp(-1/20)
#define DECAY_I 0.904837418035960f   // exp(-1/10)
#define DECAY_B 0.993079659461990f   // exp(-1/144)

// ---------------- preprocessing kernels ----------------

__global__ void k_zero_ring(_Float16* ring, int n) {
    int i = blockIdx.x * blockDim.x + threadIdx.x;
    if (i < n) ring[i] = (_Float16)0.0f;
}

__global__ void k_zero_cnt(unsigned* cnt) { *cnt = 0u; }

// WT[g][n][k] = |w_rec[g][k][n]| * sign(k) * (k!=n), f16, k padded to KP
__global__ void k_build_wrec(const float* __restrict__ w_rec, _Float16* __restrict__ WT) {
    int idx = blockIdx.x * blockDim.x + threadIdx.x;
    if (idx >= 2 * NN * KP) return;
    int k = idx & (KP - 1);
    int rest = idx >> 11;              // /KP
    int n = rest % NN;
    int g = rest / NN;
    float eff = 0.0f;
    if (k < NN && k != n) {
        float w = w_rec[((size_t)g * NN + k) * NN + n];
        float s = (k < NEXC) ? 1.0f : -1.0f;
        eff = fabsf(w) * s;
    }
    WT[idx] = (_Float16)eff;
}

// WTin[n][i] = |w_in[i][n]| * sign(i), f16, i padded to IP
__global__ void k_build_win(const float* __restrict__ w_in, _Float16* __restrict__ WTin) {
    int idx = blockIdx.x * blockDim.x + threadIdx.x;
    if (idx >= NN * IP) return;
    int i = idx % IP;
    int n = idx / IP;
    float eff = 0.0f;
    if (i < II) {
        float s = (i < NEXCI) ? 1.0f : -1.0f;
        eff = fabsf(w_in[(size_t)i * NN + n]) * s;
    }
    WTin[idx] = (_Float16)eff;
}

// X16[t][b][i] = (f16)x[t][b][i], i padded to IP
__global__ void k_build_x(const float* __restrict__ x, _Float16* __restrict__ X16) {
    int idx = blockIdx.x * blockDim.x + threadIdx.x;
    if (idx >= TT * BB * IP) return;
    int i = idx % IP;
    int tb = idx / IP;
    float v = (i < II) ? x[(size_t)tb * II + i] : 0.0f;
    X16[idx] = (_Float16)v;
}

// ---------------- persistent simulation kernel ----------------

// A-fragment for v_wmma_f32_16x16x32_f16 (this code's contiguous k order, matched by the B loads; not AMD's table order):
// lane L (group g=L/16, m=L%16): halves 0..15 -> K = k0+16g+h
__device__ __forceinline__ v16h ldA(const _Float16* __restrict__ row, int k0, int g) {
    v8h lo = *(const v8h*)(row + k0 + 16 * g);
    v8h hi = *(const v8h*)(row + k0 + 16 * g + 8);
    v16h r;
#pragma unroll
    for (int i = 0; i < 8; ++i) { r[i] = lo[i]; r[i + 8] = hi[i]; }
    return r;
}

__global__ __launch_bounds__(256) void rsnn_persist(
    const _Float16* __restrict__ WT0, const _Float16* __restrict__ WT1,
    const _Float16* __restrict__ WTin, const _Float16* __restrict__ X16,
    _Float16* __restrict__ ring,
    const float* __restrict__ v_rest, const float* __restrict__ beta,
    const float* __restrict__ noise, const float* __restrict__ u,
    float* __restrict__ out, unsigned* __restrict__ cnt)
{
    __shared__ float red[2][KSEG][32][8];          // partial C tiles, 8 KB

    const int lane = threadIdx.x & 31;
    const int lw   = threadIdx.x >> 5;             // 0..7
    const int tloc = lw >> 2;                      // 0..1 : tile within block
    const int seg  = lw & 3;                       // 0..3 : K segment
    const int tile = blockIdx.x * 2 + tloc;        // 0..249, no idle waves
    const int mt = tile / NTILE;                   // batch half
    const int nt = tile % NTILE;                   // output column tile
    const int m0 = mt * 16;
    const int n0 = nt * 16;
    const int g  = lane >> 4;
    const int lm = lane & 15;
    const int n  = n0 + lm;

    const _Float16* browWT0 = WT0 + ((size_t)n << 11) + (seg * CSEG * 32) + 16 * g;
    const _Float16* browWT1 = WT1 + ((size_t)n << 11) + (seg * CSEG * 32) + 16 * g;
    const _Float16* browIn  = WTin + (size_t)n * IP;
    const int kbase = seg * CSEG * 32;             // this wave's K origin

    // neuron state lives in VGPRs for the whole simulation (seg-0 waves own it)
    float vr = 0.f, bet = 0.f, dv = 0.f;
    float vS[8], bS[8], rS[8];
    if (seg == 0) {
        vr  = v_rest[n];
        bet = beta[n];
        dv  = (n < NEXC) ? DECAY_E : DECAY_I;
#pragma unroll
        for (int j = 0; j < 8; ++j) { vS[j] = vr; bS[j] = 0.f; rS[j] = 0.f; }
    }

    for (int t = 0; t < TT; ++t) {
        v8f acc0 = {};                 // rec group 0 chain
        v8f acc1 = {};                 // rec group 1 + input chain (independent)

        // input GEMM handled by segment 3
        if (seg == 3) {
            const _Float16* arow = X16 + ((size_t)t * BB + (m0 + lm)) * IP;
#pragma unroll
            for (int k0 = 0; k0 < IP; k0 += 32) {
                v16h a = ldA(arow, k0, g);
                v16h b = *(const v16h*)(browIn + k0 + 16 * g);
                acc1 = __builtin_amdgcn_wmma_f32_16x16x32_f16(false, a, false, b,
                                                              (short)0, acc1, false, false);
            }
        }

        // recurrent group 0 (delay 20): 16 compile-time chunks
        {
            const int slot = (t - 20) & (RING - 1);
            const _Float16* arow = ring + (((size_t)slot * BB + (m0 + lm)) << 11) + kbase;
#pragma unroll 4
            for (int c = 0; c < CSEG; ++c) {
                const int k0 = c * 32;
                v16h a = ldA(arow, k0, g);
                v16h b = *(const v16h*)(browWT0 + k0);
                acc0 = __builtin_amdgcn_wmma_f32_16x16x32_f16(false, a, false, b,
                                                              (short)0, acc0, false, false);
            }
        }
        // recurrent group 1 (delay 5)
        {
            const int slot = (t - 5) & (RING - 1);
            const _Float16* arow = ring + (((size_t)slot * BB + (m0 + lm)) << 11) + kbase;
#pragma unroll 4
            for (int c = 0; c < CSEG; ++c) {
                const int k0 = c * 32;
                v16h a = ldA(arow, k0, g);
                v16h b = *(const v16h*)(browWT1 + k0);
                acc1 = __builtin_amdgcn_wmma_f32_16x16x32_f16(false, a, false, b,
                                                              (short)0, acc1, false, false);
            }
        }

        // LDS partial-sum reduction across the 4 K segments
#pragma unroll
        for (int j = 0; j < 8; ++j) red[tloc][seg][lane][j] = acc0[j] + acc1[j];
        __syncthreads();

        if (seg == 0) {
            float tot[8];
#pragma unroll
            for (int j = 0; j < 8; ++j)
                tot[j] = red[tloc][0][lane][j] + red[tloc][1][lane][j]
                       + red[tloc][2][lane][j] + red[tloc][3][lane][j];

            // fused neuron dynamics; C layout: lane holds col n, rows m0+8g+j
            _Float16* zrow = ring + (((size_t)(t & (RING - 1)) * BB) << 11);
#pragma unroll
            for (int j = 0; j < 8; ++j) {
                const int bidx = m0 + 8 * g + j;
                const size_t tix = (size_t)t * BB * NN + (size_t)bidx * NN + n;

                float v  = vS[j];
                float bb = bS[j];
                float r  = rS[j];

                float thr_a = 0.1f + bet * bb;                        // adaptive threshold
                v = dv * v + (1.0f - dv) * vr + tot[j] + 0.01f * noise[tix];
                float p = 1.0f / (1.0f + __expf(-150.0f * (v - thr_a)));
                float z = ((u[tix] < p) && (r <= 0.0f)) ? 1.0f : 0.0f;
                r = (z > 0.0f) ? 4.0f : fmaxf(r - 1.0f, 0.0f);
                v = v - z * 0.1f;                                     // soft reset
                bb = DECAY_B * bb + (1.0f - DECAY_B) * z;             // adaptation

                vS[j] = v; bS[j] = bb; rS[j] = r;
                out[tix] = z;
                zrow[((size_t)bidx << 11) + n] = (_Float16)z;
            }
        }

        // ---- device-wide barrier: release stores, count blocks, acquire ----
        __threadfence();          // publish spikes (and order red reads before reuse)
        __syncthreads();
        if (threadIdx.x == 0) {
            __hip_atomic_fetch_add(cnt, 1u, __ATOMIC_RELEASE, __HIP_MEMORY_SCOPE_AGENT);
            const unsigned target = (unsigned)(t + 1) * (unsigned)GRIDB;
            while (__hip_atomic_load(cnt, __ATOMIC_ACQUIRE, __HIP_MEMORY_SCOPE_AGENT) < target)
                __builtin_amdgcn_s_sleep(8);
        }
        __syncthreads();
        __threadfence();          // invalidate stale lines before reading new spikes
    }
}

// ---------------- launcher ----------------

extern "C" void kernel_launch(void* const* d_in, const int* in_sizes, int n_in,
                              void* d_out, int out_size, void* d_ws, size_t ws_size,
                              hipStream_t stream) {
    const float* x      = (const float*)d_in[0];   // [T,B,I]
    const float* w_in   = (const float*)d_in[1];   // [I,N]
    const float* w_rec  = (const float*)d_in[2];   // [2,N,N]
    const float* v_rest = (const float*)d_in[3];   // [N]
    const float* beta   = (const float*)d_in[4];   // [N]
    const float* noise  = (const float*)d_in[5];   // [T,B,N]
    const float* u      = (const float*)d_in[6];   // [T,B,N]
    float* out = (float*)d_out;                    // [T,B,N]

    // workspace layout
    _Float16* WT0  = (_Float16*)d_ws;                       // [N][KP]
    _Float16* WT1  = WT0 + (size_t)NN * KP;                 // [N][KP]
    _Float16* WTin = WT1 + (size_t)NN * KP;                 // [N][IP]
    _Float16* X16  = WTin + (size_t)NN * IP;                // [T][B][IP]
    _Float16* ring = X16 + (size_t)TT * BB * IP;            // [RING][B][KP]
    unsigned* cnt  = (unsigned*)(ring + (size_t)RING * BB * KP);

    const int thr = 256;
    {
        int n = RING * BB * KP;
        k_zero_ring<<<(n + thr - 1) / thr, thr, 0, stream>>>(ring, n);
    }
    k_zero_cnt<<<1, 1, 0, stream>>>(cnt);
    {
        int n = 2 * NN * KP;
        k_build_wrec<<<(n + thr - 1) / thr, thr, 0, stream>>>(w_rec, WT0);
    }
    {
        int n = NN * IP;
        k_build_win<<<(n + thr - 1) / thr, thr, 0, stream>>>(w_in, WTin);
    }
    {
        int n = TT * BB * IP;
        k_build_x<<<(n + thr - 1) / thr, thr, 0, stream>>>(x, X16);
    }

    // one persistent kernel runs all 500 steps; 125 blocks x 8 waves,
    // 2 output tiles per block x 4 K-segments, spin barrier between steps
    rsnn_persist<<<GRIDB, 256, 0, stream>>>(WT0, WT1, WTin, X16, ring,
                                            v_rest, beta, noise, u, out, cnt);
}
